// FModule_55740085567945
// MI455X (gfx1250) — matrix operand order repaired and checked by reading, not compiled, not run on hardware
//
#include <hip/hip_runtime.h>

// ---------------------------------------------------------------------------
// MI455X (gfx1250) implementation of the PWC "FModule" reference.
//
// Heavy math (4 dense-conv decoder blocks, ~100 GFLOP) runs as implicit GEMM
// on v_wmma_f32_16x16x32_f16 (f16 A/B, f32 accumulate).  Activations live in
// a per-level NHWC f16 buffer whose channel layout makes the dense-block
// concatenations free (each conv input is a contiguous channel suffix).
// Each wave computes a 16-pixel x 32-Cout tile (two accumulators) so one A
// chunk feeds two WMMAs; halo lanes read from a zeroed ghost row so the inner
// loop is branch-free (no exec masking, no zero-fill movs, clean clauses).
// ---------------------------------------------------------------------------

typedef __attribute__((ext_vector_type(16))) _Float16 v16h;
typedef __attribute__((ext_vector_type(8)))  _Float16 v8h;
typedef __attribute__((ext_vector_type(8)))  float    v8f;

#define LRELU_NEG 0.1f

// ---------------- layout transforms ----------------------------------------

__global__ void nchw2nhwc_k(const float* __restrict__ s, float* __restrict__ d,
                            int C, int H, int W, int Bn) {
  long long idx = (long long)blockIdx.x * blockDim.x + threadIdx.x;
  long long tot = (long long)Bn * C * H * W;
  if (idx >= tot) return;
  int c = idx % C;
  long long p = idx / C;               // pixel index (b*H+y)*W+x
  int x = p % W; int y = (p / W) % H; int b = p / ((long long)W * H);
  d[idx] = s[(((long long)b * C + c) * H + y) * W + x];
}

__global__ void nchw2x_k(const float* __restrict__ s, _Float16* __restrict__ X,
                         int ldx, int off, int C, int H, int W, int Bn) {
  long long idx = (long long)blockIdx.x * blockDim.x + threadIdx.x;
  long long tot = (long long)Bn * C * H * W;
  if (idx >= tot) return;
  int x = idx % W; int y = (idx / W) % H;
  int c = (idx / ((long long)W * H)) % C; int b = idx / ((long long)W * H * C);
  long long p = ((long long)b * H + y) * W + x;
  X[p * ldx + off + c] = (_Float16)s[idx];
}

__global__ void padzero_k(_Float16* __restrict__ X, int ldx, int start, int len, int Np) {
  long long idx = (long long)blockIdx.x * blockDim.x + threadIdx.x;
  long long tot = (long long)Np * len;
  if (idx >= tot) return;
  long long p = idx / len; int j = idx % len;
  X[p * ldx + start + j] = (_Float16)0.f;
}

__global__ void zero_fill_h_k(_Float16* __restrict__ p, int n) {
  int i = blockIdx.x * blockDim.x + threadIdx.x;
  if (i < n) p[i] = (_Float16)0.f;
}

// ---------------- correlation (81 disp, max_disp=4), writes f16 into X ------

__global__ void corr_k(const float* __restrict__ A, const float* __restrict__ Bf,
                       _Float16* __restrict__ X, int ldx, int off,
                       int C, int H, int W) {
  long long p = blockIdx.x;                     // pixel index
  int d = threadIdx.x;
  int x = p % W; int y = (p / W) % H; int b = p / ((long long)W * H);
  if (d < 81) {
    int dy = d / 9 - 4, dx = d % 9 - 4;
    int yy = y + dy, xx = x + dx;
    float s = 0.f;
    if ((unsigned)yy < (unsigned)H && (unsigned)xx < (unsigned)W) {
      const float* ap = A + p * (long long)C;
      const float* bp = Bf + (((long long)b * H + yy) * W + xx) * (long long)C;
      for (int c = 0; c < C; ++c) s += ap[c] * bp[c];
    }
    s = s > 0.f ? s : LRELU_NEG * s;
    X[p * ldx + off + d] = (_Float16)s;
  }
}

// ---------------- bilinear backward warp (reference semantics) --------------

__global__ void warp_k(const float* __restrict__ feat, const float* __restrict__ flow,
                       float scale, float* __restrict__ wf,
                       int C, int H, int W, int Bn) {
  long long p = (long long)blockIdx.x * blockDim.x + threadIdx.x;
  long long tot = (long long)Bn * H * W;
  if (p >= tot) return;
  int x = p % W; int y = (p / W) % H; int b = p / ((long long)W * H);
  long long hw = (long long)H * W;
  float gx = (float)x + flow[(long long)b * 2 * hw + (long long)y * W + x] * scale;
  float gy = (float)y + flow[(long long)b * 2 * hw + hw + (long long)y * W + x] * scale;
  float x0f = floorf(gx), y0f = floorf(gy);
  float wx1 = gx - x0f, wy1 = gy - y0f, wx0 = 1.f - wx1, wy0 = 1.f - wy1;
  int x0 = (int)x0f, y0 = (int)y0f, x1 = x0 + 1, y1 = y0 + 1;
  float m00 = (x0 >= 0 && x0 <= W - 1 && y0 >= 0 && y0 <= H - 1) ? 1.f : 0.f;
  float m01 = (x1 >= 0 && x1 <= W - 1 && y0 >= 0 && y0 <= H - 1) ? 1.f : 0.f;
  float m10 = (x0 >= 0 && x0 <= W - 1 && y1 >= 0 && y1 <= H - 1) ? 1.f : 0.f;
  float m11 = (x1 >= 0 && x1 <= W - 1 && y1 >= 0 && y1 <= H - 1) ? 1.f : 0.f;
  int cx0 = x0 < 0 ? 0 : (x0 > W - 1 ? W - 1 : x0);
  int cx1 = x1 < 0 ? 0 : (x1 > W - 1 ? W - 1 : x1);
  int cy0 = y0 < 0 ? 0 : (y0 > H - 1 ? H - 1 : y0);
  int cy1 = y1 < 0 ? 0 : (y1 > H - 1 ? H - 1 : y1);
  float mask = m00 * wy0 * wx0 + m01 * wy0 * wx1 + m10 * wy1 * wx0 + m11 * wy1 * wx1;
  float keep = (mask >= 0.9999f) ? 1.f : 0.f;
  float w00 = wy0 * wx0 * m00 * keep, w01 = wy0 * wx1 * m01 * keep;
  float w10 = wy1 * wx0 * m10 * keep, w11 = wy1 * wx1 * m11 * keep;
  long long i00 = (long long)cy0 * W + cx0, i01 = (long long)cy0 * W + cx1;
  long long i10 = (long long)cy1 * W + cx0, i11 = (long long)cy1 * W + cx1;
  const float* fb = feat + (long long)b * C * hw;
  float* o = wf + p * (long long)C;
  for (int c = 0; c < C; ++c) {
    const float* fc = fb + (long long)c * hw;
    o[c] = fc[i00] * w00 + fc[i01] * w01 + fc[i10] * w10 + fc[i11] * w11;
  }
}

// ---------------- weight packing into WMMA B-operand lane layout ------------
// B (32x16 f16): lane l (lo=l&15, hi=l>>4) element e holds K = 8*hi + e (+8 if e>=8), N = lo.
// packed[((tap*nkc + kc)*nct + ct)*512 + lane*16 + e]

__global__ void pack_w_k(const float* __restrict__ w, _Float16* __restrict__ Wp,
                         int Cin, int Cout) {
  int nkc = (Cin + 31) >> 5, nct = Cout >> 4;
  long long tot = (long long)9 * nkc * nct * 512;
  long long idx = (long long)blockIdx.x * blockDim.x + threadIdx.x;
  if (idx >= tot) return;
  int e = idx & 15;
  int lane = (idx >> 4) & 31;
  long long t = idx >> 9;
  int ct = t % nct; t /= nct;
  int kc = t % nkc; int tap = t / nkc;
  int lo = lane & 15, hi = lane >> 4;
  int k = kc * 32 + 8 * hi + e + (e >= 8 ? 8 : 0);
  int n = ct * 16 + lo;
  float v = (k < Cin) ? w[((long long)tap * Cin + k) * Cout + n] : 0.f;
  Wp[idx] = (_Float16)v;
}

// ---------------- 3x3 conv as implicit GEMM on v_wmma_f32_16x16x32_f16 ------
// One wave computes a 16-pixel (x-run) x 32-Cout tile (two accumulators); the
// K loop covers 9 taps x Cin/32 chunks.  One A chunk (two b128 loads in the
// ISA 16x32 f16 lane layout) feeds two back-to-back WMMAs.  Halo lanes point
// at a zeroed ghost row so the loop body has no predication.  ldx is chosen
// so off_in + Kpad == ldx for every conv: padded-K reads hit the zeroed pad
// tail and never cross into the next pixel.

__global__ void __launch_bounds__(128)
conv3x3_wmma_k(const _Float16* __restrict__ X, _Float16* __restrict__ Xo,
               const _Float16* __restrict__ Wp, const float* __restrict__ bias,
               const _Float16* __restrict__ zrow,
               int ldx, int off_in, int Cin, int Cout, int off_out,
               int H, int W, int Bn) {
  const int nct = Cout >> 4;                    // 16-wide column tiles
  const int nct2 = Cout >> 5;                   // 32-wide column tiles per wave
  const int tilesX = W >> 4;
  const int ntiles = Bn * H * tilesX * nct2;
  int wid = blockIdx.x * (blockDim.x >> 5) + (threadIdx.x >> 5);
  if (wid >= ntiles) return;                    // wave-uniform: EXEC stays full
  int ct2 = wid % nct2;
  int pt = wid / nct2;
  int tx = pt % tilesX;
  int rem = pt / tilesX;
  int y = rem % H;
  int b = rem / H;
  int lane = threadIdx.x & 31;
  int lo = lane & 15, hi = lane >> 4;
  int nkc = (Cin + 31) >> 5;
  int px = tx * 16 + lo;                        // this lane's A row (pixel M=lo)
  const long long bstride = (long long)nct * 512;   // packed-B K-chunk stride
  v8f acc0 = {}, acc1 = {};
  for (int ky = 0; ky < 3; ++ky) {
    int yy = y + ky - 1;
    bool rok = (unsigned)yy < (unsigned)H;
    for (int kx = 0; kx < 3; ++kx) {
      int xx = px + kx - 1;
      bool ok = rok && ((unsigned)xx < (unsigned)W);
      // Halo lanes read zeros from the ghost row -> no predication in K loop.
      const _Float16* arow = ok
          ? X + (((long long)b * H + yy) * W + xx) * (long long)ldx + off_in + 8 * hi
          : zrow + 8 * hi;
      int tap = ky * 3 + kx;
      const _Float16* wrow =
          Wp + ((long long)tap * nkc * nct + 2 * ct2) * 512 + lane * 16;
      for (int kc = 0; kc < nkc; ++kc) {
        v8h a0 = *(const v8h*)(arow);           // K[8hi .. +8)
        v8h a1 = *(const v8h*)(arow + 16);      // K[16+8hi .. +8)
        v16h a = __builtin_shufflevector(a0, a1, 0, 1, 2, 3, 4, 5, 6, 7,
                                         8, 9, 10, 11, 12, 13, 14, 15);
        v16h b0 = *(const v16h*)(wrow);
        v16h b1 = *(const v16h*)(wrow + 512);
        acc0 = __builtin_amdgcn_wmma_f32_16x16x32_f16(
            false, a, false, b0, (short)0, acc0, false, false);
        acc1 = __builtin_amdgcn_wmma_f32_16x16x32_f16(
            false, a, false, b1, (short)0, acc1, false, false);
        arow += 32;
        wrow += bstride;
      }
    }
  }
  // D layout: VGPR r -> M = r + 8*hi, N = lo; columns ct2*32+lo and +16.
  float bv0 = bias[ct2 * 32 + lo];
  float bv1 = bias[ct2 * 32 + 16 + lo];
  long long obase =
      (((long long)b * H + y) * W + tx * 16) * (long long)ldx + off_out + ct2 * 32 + lo;
  for (int r = 0; r < 8; ++r) {
    int m = r + 8 * hi;
    float v0 = acc0[r] + bv0;
    float v1 = acc1[r] + bv1;
    v0 = v0 > 0.f ? v0 : LRELU_NEG * v0;        // dense-block LeakyReLU
    v1 = v1 > 0.f ? v1 : LRELU_NEG * v1;
    Xo[obase + (long long)m * ldx] = (_Float16)v0;
    Xo[obase + (long long)m * ldx + 16] = (_Float16)v1;
  }
}

// ---------------- final 3x3 conv to 2 channels (flow), scalar ---------------

__global__ void conv_pw_k(const _Float16* __restrict__ X, int ldx, int Cin,
                          const float* __restrict__ w, const float* __restrict__ bias,
                          float* __restrict__ flow, int H, int W, int Bn) {
  long long t = (long long)blockIdx.x * blockDim.x + threadIdx.x;
  long long tot = (long long)Bn * H * W * 2;
  if (t >= tot) return;
  int co = t & 1;
  long long p = t >> 1;
  int x = p % W; int y = (p / W) % H; int b = p / ((long long)W * H);
  float s = bias[co];
  for (int ky = 0; ky < 3; ++ky) {
    int yy = y + ky - 1; if ((unsigned)yy >= (unsigned)H) continue;
    for (int kx = 0; kx < 3; ++kx) {
      int xx = x + kx - 1; if ((unsigned)xx >= (unsigned)W) continue;
      const _Float16* xr = X + (((long long)b * H + yy) * W + xx) * (long long)ldx;
      const float* wr = w + ((long long)(ky * 3 + kx) * Cin) * 2 + co;
      for (int ci = 0; ci < Cin; ++ci) s += (float)xr[ci] * wr[2 * ci];
    }
  }
  flow[(((long long)b * 2 + co) * H + y) * W + x] = s;
}

// ---------------- ConvTranspose2d(k=4, s=2, p=1) variants -------------------

__global__ void deconv4_flow_k(const float* __restrict__ xin, const float* __restrict__ w,
                               const float* __restrict__ bias, float* __restrict__ out,
                               int H, int W, int Bn) {
  int Ho = 2 * H, Wo = 2 * W;
  long long t = (long long)blockIdx.x * blockDim.x + threadIdx.x;
  long long tot = (long long)Bn * 2 * Ho * Wo;
  if (t >= tot) return;
  int ox = t % Wo; int oy = (t / Wo) % Ho;
  int co = (t / ((long long)Wo * Ho)) % 2; int b = t / ((long long)Wo * Ho * 2);
  float s = bias[co];
  for (int ky = 0; ky < 4; ++ky) {
    int dy = oy + ky - 2; if (dy & 1) continue; int iy = dy >> 1;
    if ((unsigned)iy >= (unsigned)H) continue;
    for (int kx = 0; kx < 4; ++kx) {
      int dx = ox + kx - 2; if (dx & 1) continue; int ix = dx >> 1;
      if ((unsigned)ix >= (unsigned)W) continue;
      for (int ci = 0; ci < 2; ++ci)
        s += xin[(((long long)b * 2 + ci) * H + iy) * W + ix] *
             w[((ky * 4 + kx) * 2 + ci) * 2 + co];
    }
  }
  out[(((long long)b * 2 + co) * Ho + oy) * Wo + ox] = s;
}

__global__ void deconv4_feat_k(const _Float16* __restrict__ X, int ldx, int Ci,
                               int H, int W, const float* __restrict__ w,
                               const float* __restrict__ bias, float* __restrict__ out,
                               int Bn) {
  int Ho = 2 * H, Wo = 2 * W;
  long long t = (long long)blockIdx.x * blockDim.x + threadIdx.x;
  long long tot = (long long)Bn * 2 * Ho * Wo;
  if (t >= tot) return;
  int ox = t % Wo; int oy = (t / Wo) % Ho;
  int co = (t / ((long long)Wo * Ho)) % 2; int b = t / ((long long)Wo * Ho * 2);
  float s = bias[co];
  for (int ky = 0; ky < 4; ++ky) {
    int dy = oy + ky - 2; if (dy & 1) continue; int iy = dy >> 1;
    if ((unsigned)iy >= (unsigned)H) continue;
    for (int kx = 0; kx < 4; ++kx) {
      int dx = ox + kx - 2; if (dx & 1) continue; int ix = dx >> 1;
      if ((unsigned)ix >= (unsigned)W) continue;
      const _Float16* xr = X + (((long long)b * H + iy) * W + ix) * (long long)ldx;
      const float* wr = w + ((long long)(ky * 4 + kx) * Ci) * 2 + co;
      for (int ci = 0; ci < Ci; ++ci) s += (float)xr[ci] * wr[2 * ci];
    }
  }
  out[(((long long)b * 2 + co) * Ho + oy) * Wo + ox] = s;
}

// ---------------- output assembly -------------------------------------------

__global__ void write_cat_diff_k(const float* __restrict__ f1, const float* __restrict__ wf,
                                 float* __restrict__ out, int C, int H, int W, int Bn) {
  long long idx = (long long)blockIdx.x * blockDim.x + threadIdx.x;
  long long tot = (long long)Bn * C * H * W;
  if (idx >= tot) return;
  int x = idx % W; int y = (idx / W) % H;
  int c = (idx / ((long long)W * H)) % C; int b = idx / ((long long)W * H * C);
  float v = f1[idx];
  long long ob = (((long long)b * 2 * C + c) * H + y) * W + x;
  out[ob] = v;
  out[ob + (long long)C * H * W] =
      v - wf[(((long long)b * H + y) * W + x) * (long long)C + c];
}

__global__ void write_cat2_k(const float* __restrict__ f1, const float* __restrict__ f2,
                             float* __restrict__ out, int C, int H, int W, int Bn) {
  long long idx = (long long)blockIdx.x * blockDim.x + threadIdx.x;
  long long tot = (long long)Bn * C * H * W;
  if (idx >= tot) return;
  int x = idx % W; int y = (idx / W) % H;
  int c = (idx / ((long long)W * H)) % C; int b = idx / ((long long)W * H * C);
  long long ob = (((long long)b * 2 * C + c) * H + y) * W + x;
  out[ob] = f1[idx];
  out[ob + (long long)C * H * W] = f1[idx] - f2[idx];
}

// ---------------------------------------------------------------------------
// Host orchestration
// ---------------------------------------------------------------------------

struct LevelCfg { int H, W, C, cin, ldx, pbase; float scale; };

extern "C" void kernel_launch(void* const* d_in, const int* in_sizes, int n_in,
                              void* d_out, int out_size, void* d_ws, size_t ws_size,
                              hipStream_t stream) {
  (void)in_sizes; (void)n_in; (void)out_size; (void)ws_size;

  // Inputs (setup_inputs insertion order): feat1 L1..L6, feat2 L1..L6, params.
  const float* F1[6]; const float* F2[6];
  for (int i = 0; i < 6; ++i) { F1[i] = (const float*)d_in[i]; F2[i] = (const float*)d_in[6 + i]; }
  float* out = (float*)d_out;

  // Workspace layout (needs ~64 MB; working set is L2-resident on MI455X).
  char* ws = (char*)d_ws;
  _Float16* X  = (_Float16*)(ws);                  // 36 MB dense NHWC f16 buffer
  _Float16* Wp = (_Float16*)(ws + (36ull << 20));  //  2 MB packed weights
  float* Anh   = (float*)(ws + (38ull << 20));     //  8 MB f1 NHWC f32
  float* WF    = (float*)(ws + (46ull << 20));     // 15 MB warped f2 / f2 NHWC
  float* UF    = (float*)(ws + (61ull << 20));     //  1 MB upflow (NCHW)
  float* UX    = (float*)(ws + (62ull << 20));     //  1 MB upfeat (NCHW)
  float* FL    = (float*)(ws + (63ull << 20));     //  flow (NCHW)
  _Float16* ZR = (_Float16*)(ws + (63ull << 20) + (512u << 10));  // ghost zero row

  // Decode levels 6..3.  ldx = 448 + ceil(cin/32)*32 (multiple of 8 halves).
  static const LevelCfg LV[4] = {
    { 7,  16, 196,  81, 544, 12, 0.f    },   // level6
    {14,  32, 128, 213, 672, 28, 0.625f },   // level5: 20/32
    {28,  64,  96, 181, 640, 44, 1.25f  },   // level4: 20/16
    {56, 128,  64, 149, 608, 60, 2.5f   },   // level3: 20/8
  };
  static const int FI[4] = {5, 4, 3, 2};     // feat array index per decode level
  static const int DF[5]    = {128, 128, 96, 64, 32};
  static const int OFFIN[5] = {448, 320, 192, 96, 32};

  // d_out flat offsets (f32 elements), reference return order.
  const long long O1 = 7340032LL, O2 = 14680064LL, O3 = 18350080LL,
                  O4 = 19726336LL, O5 = 20185088LL;
  const long long DIFF_OFF[4] = {0, O4, O3, O2};   // per decode level 5,4,3

  const int TB = 256;
  #define NB(n) ((int)(((long long)(n) + TB - 1) / TB))

  // Zero the ghost row (1024 halves; K-chunk reads of halo lanes land here).
  zero_fill_h_k<<<4, 256, 0, stream>>>(ZR, 1024);

  for (int li = 0; li < 4; ++li) {
    const LevelCfg& L = LV[li];
    const int H = L.H, W = L.W, C = L.C, cin = L.cin, ldx = L.ldx;
    const int Np = 4 * H * W;
    const float* f1 = F1[FI[li]];
    const float* f2 = F2[FI[li]];

    // f1 -> NHWC f32 (correlation A operand)
    nchw2nhwc_k<<<NB((long long)Np * C), TB, 0, stream>>>(f1, Anh, C, H, W, 4);

    if (li == 0) {
      // level6: correlate against raw f2
      nchw2nhwc_k<<<NB((long long)Np * C), TB, 0, stream>>>(f2, WF, C, H, W, 4);
    } else {
      // warp f2 by scaled upflow; emit diff part of this level's output now
      warp_k<<<NB(Np), TB, 0, stream>>>(f2, UF, L.scale, WF, C, H, W, 4);
      write_cat_diff_k<<<NB((long long)Np * C), TB, 0, stream>>>(
          f1, WF, out + DIFF_OFF[li], C, H, W, 4);
    }

    // zero the K-pad tail [448+cin, ldx) once; all 5 GEMMs read into it
    {
      int len = ldx - (448 + cin);
      padzero_k<<<NB((long long)Np * len), TB, 0, stream>>>(X, ldx, 448 + cin, len, Np);
    }

    // cost volume -> X channels [448, 529)
    corr_k<<<Np, 96, 0, stream>>>(Anh, WF, X, ldx, 448, C, H, W);

    if (li > 0) {
      // concat [corr, a, upflow, upfeat] into x0 region
      nchw2x_k<<<NB((long long)Np * C), TB, 0, stream>>>(f1, X, ldx, 448 + 81, C, H, W, 4);
      nchw2x_k<<<NB((long long)Np * 2), TB, 0, stream>>>(UF, X, ldx, 448 + 81 + C, 2, H, W, 4);
      nchw2x_k<<<NB((long long)Np * 2), TB, 0, stream>>>(UX, X, ldx, 448 + 81 + C + 2, 2, H, W, 4);
    }

    // five dense WMMA convs; each writes its 32-multiple Cout block in front
    for (int i = 0; i < 5; ++i) {
      int Cin = 448 + cin - OFFIN[i];
      int Cout = DF[i];
      const float* w  = (const float*)d_in[L.pbase + 2 * i];
      const float* bs = (const float*)d_in[L.pbase + 2 * i + 1];
      int nkc = (Cin + 31) / 32, nct = Cout / 16;
      long long ptot = (long long)9 * nkc * nct * 512;
      pack_w_k<<<NB(ptot), TB, 0, stream>>>(w, Wp, Cin, Cout);
      int ntiles = 4 * H * (W / 16) * (Cout / 32);
      conv3x3_wmma_k<<<(ntiles + 3) / 4, 128, 0, stream>>>(
          X, X, Wp, bs, ZR, ldx, OFFIN[i], Cin, Cout, OFFIN[i] - Cout, H, W, 4);
    }

    // flow head
    {
      const float* w  = (const float*)d_in[L.pbase + 10];
      const float* bs = (const float*)d_in[L.pbase + 11];
      conv_pw_k<<<NB((long long)Np * 2), TB, 0, stream>>>(
          X, ldx, 448 + cin, w, bs, FL, H, W, 4);
    }

    if (li < 3) {
      const float* ufw = (const float*)d_in[L.pbase + 12];
      const float* ufb = (const float*)d_in[L.pbase + 13];
      const float* uxw = (const float*)d_in[L.pbase + 14];
      const float* uxb = (const float*)d_in[L.pbase + 15];
      long long tot = (long long)4 * 2 * (2 * H) * (2 * W);
      deconv4_flow_k<<<NB(tot), TB, 0, stream>>>(FL, ufw, ufb, UF, H, W, 4);
      deconv4_feat_k<<<NB(tot), TB, 0, stream>>>(X, ldx, 448 + cin, H, W, uxw, uxb, UX, 4);
    } else {
      // final_up = deconv(flow_level3, up_w); level2 diff + concat
      const float* upw = (const float*)d_in[72];
      const float* upb = (const float*)d_in[73];
      long long tot = (long long)4 * 2 * 112 * 256;
      deconv4_flow_k<<<NB(tot), TB, 0, stream>>>(FL, upw, upb, UF, 56, 128, 4);
      int Np2 = 4 * 112 * 256;
      warp_k<<<NB(Np2), TB, 0, stream>>>(F2[1], UF, 1.f, WF, 32, 112, 256, 4);
      write_cat_diff_k<<<NB((long long)Np2 * 32), TB, 0, stream>>>(
          F1[1], WF, out + O1, 32, 112, 256, 4);
    }
  }

  // out5: [f1_l6, f1_l6 - f2_l6]
  write_cat2_k<<<NB((long long)4 * 196 * 7 * 16), TB, 0, stream>>>(
      F1[5], F2[5], out + O5, 196, 7, 16, 4);

  // out0: plain copy of feat1_level1
  (void)hipMemcpyAsync(out, F1[0], sizeof(float) * 7340032ull,
                       hipMemcpyDeviceToDevice, stream);
  #undef NB
}
